// GINLayer_27934467293294
// MI455X (gfx1250) — matrix operand order repaired and checked by reading, not compiled, not run on hardware
//
#include <hip/hip_runtime.h>

#define D 128
#define BN_EPS 1e-5f

typedef __attribute__((ext_vector_type(16))) _Float16 v16h;
typedef __attribute__((ext_vector_type(8)))  float    v8f;

__device__ __forceinline__ void atomicAddF(float* p, float v) {
    // relaxed, agent scope -> hardware global_atomic_add_f32 (L2-resident working set)
    __hip_atomic_fetch_add(p, v, __ATOMIC_RELAXED, __HIP_MEMORY_SCOPE_AGENT);
}

// ---------------------------------------------------------------------------
// Kernel 1: h0 = (2+eps)*x   (self-loop + (1+eps)x folded together),
// convert W1/W2 to f16 for WMMA, zero BN accumulators.
// ---------------------------------------------------------------------------
__global__ __launch_bounds__(256) void gin_prep(const float* __restrict__ x,
                                                const float* __restrict__ epsp,
                                                const float* __restrict__ W1,
                                                const float* __restrict__ W2,
                                                float* __restrict__ h0,
                                                _Float16* __restrict__ W1h,
                                                _Float16* __restrict__ W2h,
                                                float* __restrict__ stats,
                                                long long total) {
    long long i = (long long)blockIdx.x * 256 + threadIdx.x;
    if (i < total) h0[i] = (2.0f + epsp[0]) * x[i];
    if (i < D * D) {
        W1h[i] = (_Float16)W1[i];
        W2h[i] = (_Float16)W2[i];
    }
    if (i < 512) stats[i] = 0.0f;   // sums[128] | sumsq[128] | scale[128] | shift[128]
}

// ---------------------------------------------------------------------------
// Kernel 2: symmetric edge scatter. One thread = (edge, 4-feature quad).
// h0[dst] += x[src]; h0[src] += x[dst]
// ---------------------------------------------------------------------------
__global__ __launch_bounds__(256) void gin_scatter(const float* __restrict__ x,
                                                   const int* __restrict__ ei,
                                                   float* __restrict__ h0, int E) {
    long long gid = (long long)blockIdx.x * 256 + threadIdx.x;
    if (gid >= (long long)E * 32) return;
    int e = (int)(gid >> 5);
    int q = ((int)gid & 31) * 4;
    int s = ei[e];
    int d = ei[E + e];
    float4 xs = *(const float4*)(x + (size_t)s * D + q);
    float4 xd = *(const float4*)(x + (size_t)d * D + q);
    float* pd = h0 + (size_t)d * D + q;
    float* ps = h0 + (size_t)s * D + q;
    atomicAddF(pd + 0, xs.x); atomicAddF(pd + 1, xs.y);
    atomicAddF(pd + 2, xs.z); atomicAddF(pd + 3, xs.w);
    atomicAddF(ps + 0, xd.x); atomicAddF(ps + 1, xd.y);
    atomicAddF(ps + 2, xd.z); atomicAddF(ps + 3, xd.w);
}

// ---------------------------------------------------------------------------
// A-fragment (16x32 f16), contiguous k order: lane = M + 16*kHalf.
//   VGPR j holds k = kHalf*16 + j*2 (+0,+1)
// => per lane one contiguous 16-float run at k = hi*16.
// B-fragment mirrors A with N as the striped row: 16 contiguous halves/lane.
// ---------------------------------------------------------------------------
__device__ __forceinline__ v16h pack_a(float4 q0, float4 q1, float4 q2, float4 q3) {
    v16h a;
    a[0] = (_Float16)q0.x; a[1] = (_Float16)q0.y; a[2] = (_Float16)q0.z; a[3] = (_Float16)q0.w;
    a[4] = (_Float16)q1.x; a[5] = (_Float16)q1.y; a[6] = (_Float16)q1.z; a[7] = (_Float16)q1.w;
    a[8]  = (_Float16)q2.x; a[9]  = (_Float16)q2.y; a[10] = (_Float16)q2.z; a[11] = (_Float16)q2.w;
    a[12] = (_Float16)q3.x; a[13] = (_Float16)q3.y; a[14] = (_Float16)q3.z; a[15] = (_Float16)q3.w;
    return a;
}

// Stage the 32KB f16 weight matrix into LDS (once per 256-thread block):
// 2048 x 16B chunks, 8 chunks per thread -> global_load_b128 + ds_store_b128.
__device__ __forceinline__ void stage_weights(_Float16* __restrict__ Wsh,
                                              const _Float16* __restrict__ Wg) {
    const float4* src = (const float4*)Wg;
    float4*       dst = (float4*)Wsh;
#pragma unroll
    for (int j = 0; j < 8; j++) {
        int idx = threadIdx.x + 256 * j;
        dst[idx] = src[idx];
    }
    __syncthreads();
}

// ---------------------------------------------------------------------------
// Kernel 3: GEMM1  h1 = h0 @ W1^T + b1, fused per-feature sum/sumsq for BN.
// 256 threads = 8 waves; W1 staged in LDS; 1 wave = 16 rows x 128 cols.
// WMMA f32_16x16x32_f16, K=128 in 4 steps, B fragments from ds_load.
// ---------------------------------------------------------------------------
__global__ __launch_bounds__(256) void gin_gemm1(const float* __restrict__ h0,
                                                 const _Float16* __restrict__ W1h,
                                                 const float* __restrict__ b1,
                                                 float* __restrict__ h1,
                                                 float* __restrict__ sums,
                                                 float* __restrict__ sumsq,
                                                 int rowTiles) {
    __shared__ _Float16 Wsh[D * D];
    stage_weights(Wsh, W1h);

    int tile = blockIdx.x * 8 + (threadIdx.x >> 5);
    if (tile >= rowTiles) return;          // wave-uniform: EXEC all-ones for WMMA
    int lane = threadIdx.x & 31;
    int r = lane & 15, hi = lane >> 4;
    int rowBase = tile * 16;

    v8f acc[8] = {};
    const float* arow = h0 + (size_t)(rowBase + r) * D;

#pragma unroll
    for (int kt = 0; kt < 4; kt++) {
        int kb = kt * 32;
        if (kt < 3) __builtin_prefetch(arow + kb + 32, 0, 1);
        float4 q0 = *(const float4*)(arow + kb + hi * 16);
        float4 q1 = *(const float4*)(arow + kb + hi * 16 + 4);
        float4 q2 = *(const float4*)(arow + kb + hi * 16 + 8);
        float4 q3 = *(const float4*)(arow + kb + hi * 16 + 12);
        v16h a = pack_a(q0, q1, q2, q3);
#pragma unroll
        for (int nt = 0; nt < 8; nt++) {
            v16h b = *(const v16h*)(Wsh + (nt * 16 + r) * D + kb + hi * 16);
            acc[nt] = __builtin_amdgcn_wmma_f32_16x16x32_f16(
                false, a, false, b, (short)0, acc[nt], false, false);
        }
    }

#pragma unroll
    for (int nt = 0; nt < 8; nt++) {
        int col = nt * 16 + r;
        float bias = b1[col];
        float s = 0.0f, s2 = 0.0f;
#pragma unroll
        for (int v = 0; v < 8; v++) {      // C layout: M = v + 8*hi, N = r
            float val = acc[nt][v] + bias;
            h1[(size_t)(rowBase + v + hi * 8) * D + col] = val;
            s += val; s2 += val * val;
        }
        atomicAddF(sums + col, s);
        atomicAddF(sumsq + col, s2);
    }
}

// ---------------------------------------------------------------------------
// Kernel 4: BN finalize -> scale/shift per feature.
// ---------------------------------------------------------------------------
__global__ __launch_bounds__(128) void gin_bn(float* __restrict__ stats,
                                              const float* __restrict__ gamma,
                                              const float* __restrict__ beta,
                                              float invN) {
    int t = threadIdx.x;
    float mean = stats[t] * invN;
    float var  = stats[128 + t] * invN - mean * mean;
    float inv  = rsqrtf(var + BN_EPS);
    float sc   = gamma[t] * inv;
    stats[256 + t] = sc;
    stats[384 + t] = beta[t] - sc * mean;
}

// ---------------------------------------------------------------------------
// Kernel 5: GEMM2  out = relu(bn(h1)) @ W2^T + b2, BN+ReLU fused into A-pack.
// ---------------------------------------------------------------------------
__global__ __launch_bounds__(256) void gin_gemm2(const float* __restrict__ h1,
                                                 const float* __restrict__ stats,
                                                 const _Float16* __restrict__ W2h,
                                                 const float* __restrict__ b2,
                                                 float* __restrict__ out,
                                                 int rowTiles) {
    __shared__ _Float16 Wsh[D * D];
    stage_weights(Wsh, W2h);

    int tile = blockIdx.x * 8 + (threadIdx.x >> 5);
    if (tile >= rowTiles) return;
    int lane = threadIdx.x & 31;
    int r = lane & 15, hi = lane >> 4;
    int rowBase = tile * 16;
    const float* scale = stats + 256;
    const float* shift = stats + 384;

    v8f acc[8] = {};
    const float* arow = h1 + (size_t)(rowBase + r) * D;

#pragma unroll
    for (int kt = 0; kt < 4; kt++) {
        int kb = kt * 32;
        if (kt < 3) __builtin_prefetch(arow + kb + 32, 0, 1);
        int o0 = kb + hi * 16, o1 = kb + hi * 16 + 8;
        float4 q0 = *(const float4*)(arow + o0);
        float4 q1 = *(const float4*)(arow + o0 + 4);
        float4 q2 = *(const float4*)(arow + o1);
        float4 q3 = *(const float4*)(arow + o1 + 4);
        float4 s0 = *(const float4*)(scale + o0);
        float4 s1 = *(const float4*)(scale + o0 + 4);
        float4 s2 = *(const float4*)(scale + o1);
        float4 s3 = *(const float4*)(scale + o1 + 4);
        float4 t0 = *(const float4*)(shift + o0);
        float4 t1 = *(const float4*)(shift + o0 + 4);
        float4 t2 = *(const float4*)(shift + o1);
        float4 t3 = *(const float4*)(shift + o1 + 4);
        float4 g0, g1, g2, g3;
        g0.x = fmaxf(0.0f, q0.x * s0.x + t0.x); g0.y = fmaxf(0.0f, q0.y * s0.y + t0.y);
        g0.z = fmaxf(0.0f, q0.z * s0.z + t0.z); g0.w = fmaxf(0.0f, q0.w * s0.w + t0.w);
        g1.x = fmaxf(0.0f, q1.x * s1.x + t1.x); g1.y = fmaxf(0.0f, q1.y * s1.y + t1.y);
        g1.z = fmaxf(0.0f, q1.z * s1.z + t1.z); g1.w = fmaxf(0.0f, q1.w * s1.w + t1.w);
        g2.x = fmaxf(0.0f, q2.x * s2.x + t2.x); g2.y = fmaxf(0.0f, q2.y * s2.y + t2.y);
        g2.z = fmaxf(0.0f, q2.z * s2.z + t2.z); g2.w = fmaxf(0.0f, q2.w * s2.w + t2.w);
        g3.x = fmaxf(0.0f, q3.x * s3.x + t3.x); g3.y = fmaxf(0.0f, q3.y * s3.y + t3.y);
        g3.z = fmaxf(0.0f, q3.z * s3.z + t3.z); g3.w = fmaxf(0.0f, q3.w * s3.w + t3.w);
        v16h a = pack_a(g0, g1, g2, g3);
#pragma unroll
        for (int nt = 0; nt < 8; nt++) {
            v16h b = *(const v16h*)(Wsh + (nt * 16 + r) * D + kb + hi * 16);
            acc[nt] = __builtin_amdgcn_wmma_f32_16x16x32_f16(
                false, a, false, b, (short)0, acc[nt], false, false);
        }
    }

#pragma unroll
    for (int nt = 0; nt < 8; nt++) {
        int col = nt * 16 + r;
        float bias = b2[col];
#pragma unroll
        for (int v = 0; v < 8; v++) {
            out[(size_t)(rowBase + v + hi * 8) * D + col] = acc[nt][v] + bias;
        }
    }
}

// ---------------------------------------------------------------------------
// Launch
// ---------------------------------------------------------------------------
extern "C" void kernel_launch(void* const* d_in, const int* in_sizes, int n_in,
                              void* d_out, int out_size, void* d_ws, size_t ws_size,
                              hipStream_t stream) {
    (void)n_in; (void)out_size; (void)ws_size;
    const float* x     = (const float*)d_in[0];
    const int*   ei    = (const int*)d_in[1];
    const float* eps   = (const float*)d_in[2];
    const float* W1    = (const float*)d_in[3];
    const float* b1    = (const float*)d_in[4];
    const float* gamma = (const float*)d_in[5];
    const float* beta  = (const float*)d_in[6];
    const float* W2    = (const float*)d_in[7];
    const float* b2    = (const float*)d_in[8];

    int N = in_sizes[0] / D;   // 100000
    int E = in_sizes[1] / 2;   // 1600000

    // d_out doubles as h0 scratch (fully rewritten by GEMM2 at the end).
    float*    h0    = (float*)d_out;
    float*    h1    = (float*)d_ws;                       // N*D f32
    _Float16* W1h   = (_Float16*)(h1 + (size_t)N * D);    // D*D f16
    _Float16* W2h   = W1h + D * D;                        // D*D f16
    float*    stats = (float*)(W2h + D * D);              // 512 f32

    long long total = (long long)N * D;
    gin_prep<<<(int)((total + 255) / 256), 256, 0, stream>>>(
        x, eps, W1, W2, h0, W1h, W2h, stats, total);

    long long sw = (long long)E * 32;
    gin_scatter<<<(int)((sw + 255) / 256), 256, 0, stream>>>(x, ei, h0, E);

    int rowTiles = N / 16;               // 6250, N divisible by 16
    int gb = (rowTiles + 7) / 8;         // 8 waves/block, 1 row-tile per wave
    gin_gemm1<<<gb, 256, 0, stream>>>(h0, W1h, b1, h1, stats, stats + 128, rowTiles);
    gin_bn<<<1, 128, 0, stream>>>(stats, gamma, beta, 1.0f / (float)N);
    gin_gemm2<<<gb, 256, 0, stream>>>(h1, stats, W2h, b2, (float*)d_out, rowTiles);
}
